// SparseAttention_73830487818393
// MI455X (gfx1250) — matrix operand order repaired and checked by reading, not compiled, not run on hardware
//
#include <hip/hip_runtime.h>
#include <hip/hip_bf16.h>
#include <math.h>

#define B_ 4
#define S_ 2048
#define E_ 1024
#define H_ 16
#define HD_ 64
#define WIN_ 10

typedef __attribute__((ext_vector_type(16))) __bf16 v16bf;
typedef __attribute__((ext_vector_type(8)))  float  v8f;

union BF16Frag { uint4 u[2]; v16bf v; };

__device__ __forceinline__ unsigned short f2bf(float f) {
  unsigned int u = __float_as_uint(f);
  u += 0x7fffu + ((u >> 16) & 1u);           // round-to-nearest-even
  return (unsigned short)(u >> 16);
}

__device__ __forceinline__ v8f wmma_bf16(v16bf a, v16bf b, v8f c) {
  // D = A(16x32 bf16) x B(32x16 bf16) + C(16x16 f32)
  return __builtin_amdgcn_wmma_f32_16x16x32_bf16(false, a, false, b, (short)0, c, false, false);
}

__device__ __forceinline__ v8f vzero8() {
  v8f z; 
  #pragma unroll
  for (int i = 0; i < 8; ++i) z[i] = 0.f;
  return z;
}

// ---------------- fp32 -> bf16 convert ----------------
__global__ void to_bf16_kernel(const float* __restrict__ src,
                               unsigned short* __restrict__ dst, int n) {
  int i = blockIdx.x * blockDim.x + threadIdx.x;
  if (i < n) dst[i] = f2bf(src[i]);
}

// ---------------- per-batch column sum of V ----------------
__global__ void vsum_kernel(const float* __restrict__ V, float* __restrict__ Vsum) {
  int e = blockIdx.x * 256 + threadIdx.x;
  int b = blockIdx.y;
  const float* vp = V + (size_t)b * S_ * E_ + e;
  float acc = 0.f;
  for (int s = 0; s < S_; ++s) acc += vp[(size_t)s * E_];
  Vsum[b * E_ + e] = acc;
}

// ---------------- in-projection GEMM ----------------
// C[M=8192,N=1024] = X * W^T + bias  per kind (q/k/v via blockIdx.z)
// kind0 -> QPb (scaled 1/8), kind1 -> KPb, kind2 -> VTb ([B,H,hd,S] transposed)
#define LDST 40  // LDS row stride in halves (80B, 16B-aligned)

__global__ __launch_bounds__(256)
void inproj_gemm(const unsigned short* __restrict__ Qb,
                 const unsigned short* __restrict__ Kb,
                 const unsigned short* __restrict__ Vb,
                 const unsigned short* __restrict__ Wb,   // [3E,E]
                 const float* __restrict__ bias,          // [3E]
                 unsigned short* __restrict__ QPb,
                 unsigned short* __restrict__ KPb,
                 unsigned short* __restrict__ VTb) {
  const int kind = blockIdx.z;
  const unsigned short* X = (kind == 0) ? Qb : ((kind == 1) ? Kb : Vb);
  const unsigned short* W = Wb + (size_t)kind * E_ * E_;
  const int m0 = blockIdx.x * 128;
  const int n0 = blockIdx.y * 128;

  __shared__ unsigned short At[128][LDST];
  __shared__ unsigned short Bt[128][LDST];

  const int tid = threadIdx.x;
  const int lane = tid & 31, wid = tid >> 5;
  const int wm = (wid >> 1) * 32, wn = (wid & 1) * 64;
  const int half = lane >> 4, l15 = lane & 15;
  const int lrow = tid >> 1, lcol = (tid & 1) * 16;

  v8f acc[2][4];
  #pragma unroll
  for (int mi = 0; mi < 2; ++mi)
    #pragma unroll
    for (int ni = 0; ni < 4; ++ni) acc[mi][ni] = vzero8();

  for (int k0 = 0; k0 < E_; k0 += 32) {
    const uint4* ga = (const uint4*)(X + (size_t)(m0 + lrow) * E_ + k0 + lcol);
    uint4 a0 = ga[0], a1 = ga[1];
    const uint4* gb = (const uint4*)(W + (size_t)(n0 + lrow) * E_ + k0 + lcol);
    uint4 b0 = gb[0], b1 = gb[1];
    if (k0 + 32 < E_) {  // gfx1250 global_prefetch of next K-tile
      __builtin_prefetch(X + (size_t)(m0 + lrow) * E_ + k0 + 32 + lcol, 0, 3);
      __builtin_prefetch(W + (size_t)(n0 + lrow) * E_ + k0 + 32 + lcol, 0, 3);
    }
    __syncthreads();
    *(uint4*)&At[lrow][lcol]     = a0;
    *(uint4*)&At[lrow][lcol + 8] = a1;
    *(uint4*)&Bt[lrow][lcol]     = b0;
    *(uint4*)&Bt[lrow][lcol + 8] = b1;
    __syncthreads();

    BF16Frag afr[2], bfr[4];
    #pragma unroll
    for (int mi = 0; mi < 2; ++mi) {
      int r = wm + mi * 16 + l15;
      afr[mi].u[0] = *(const uint4*)&At[r][half * 8];
      afr[mi].u[1] = *(const uint4*)&At[r][16 + half * 8];
    }
    #pragma unroll
    for (int ni = 0; ni < 4; ++ni) {
      int c = wn + ni * 16 + l15;
      bfr[ni].u[0] = *(const uint4*)&Bt[c][half * 8];
      bfr[ni].u[1] = *(const uint4*)&Bt[c][16 + half * 8];
    }
    #pragma unroll
    for (int mi = 0; mi < 2; ++mi)
      #pragma unroll
      for (int ni = 0; ni < 4; ++ni)
        acc[mi][ni] = wmma_bf16(afr[mi].v, bfr[ni].v, acc[mi][ni]);
  }

  #pragma unroll
  for (int mi = 0; mi < 2; ++mi)
    #pragma unroll
    for (int ni = 0; ni < 4; ++ni) {
      int col = n0 + wn + ni * 16 + l15;
      float bc = bias[kind * E_ + col];
      #pragma unroll
      for (int r = 0; r < 8; ++r) {
        int row = m0 + wm + mi * 16 + r + half * 8;
        float v = acc[mi][ni][r] + bc;
        if (kind == 0) {
          QPb[(size_t)row * E_ + col] = f2bf(v * 0.125f);       // * hd^-0.5
        } else if (kind == 1) {
          KPb[(size_t)row * E_ + col] = f2bf(v);
        } else {
          int bb = row >> 11, s = row & (S_ - 1);
          int h = col >> 6, d = col & (HD_ - 1);
          VTb[(((size_t)bb * H_ + h) * HD_ + d) * S_ + s] = f2bf(v);
        }
      }
    }
}

// ---------------- flash attention (global MHA core) ----------------
__global__ __launch_bounds__(128)
void flash_attn(const unsigned short* __restrict__ QPb,
                const unsigned short* __restrict__ KPb,
                const unsigned short* __restrict__ VTb,
                unsigned short* __restrict__ AOb) {
  const int qt = blockIdx.x, h = blockIdx.y, b = blockIdx.z;
  const int lane = threadIdx.x & 31, wid = threadIdx.x >> 5;
  const int half = lane >> 4, l15 = lane & 15;

  __shared__ unsigned short P[4][16][72];   // per-wave P tile, stride 144B

  // Q fragments (A-operand): 16 q-rows x 64 hd, 2 K-steps of 32
  const int qrow = qt * 64 + wid * 16 + l15;
  BF16Frag qf[2];
  {
    const unsigned short* qp = QPb + ((size_t)b * S_ + qrow) * E_ + h * HD_;
    #pragma unroll
    for (int ks = 0; ks < 2; ++ks) {
      qf[ks].u[0] = *(const uint4*)(qp + ks * 32 + half * 8);
      qf[ks].u[1] = *(const uint4*)(qp + ks * 32 + 16 + half * 8);
    }
  }

  v8f acc_o[4];
  #pragma unroll
  for (int nv = 0; nv < 4; ++nv) acc_o[nv] = vzero8();
  float mrow[8], lrow[8];
  #pragma unroll
  for (int r = 0; r < 8; ++r) { mrow[r] = -1e30f; lrow[r] = 0.f; }

  const unsigned short* kbase = KPb + (size_t)b * S_ * E_ + h * HD_;
  const unsigned short* vbase = VTb + ((size_t)b * H_ + h) * HD_ * S_;

  for (int kt = 0; kt < S_ / 64; ++kt) {
    // ---- scores: S = Qh * Kh^T  (16 x 64) ----
    v8f sc[4];
    #pragma unroll
    for (int ni = 0; ni < 4; ++ni) sc[ni] = vzero8();
    #pragma unroll
    for (int ni = 0; ni < 4; ++ni) {
      const unsigned short* kp = kbase + (size_t)(kt * 64 + ni * 16 + l15) * E_;
      #pragma unroll
      for (int ks = 0; ks < 2; ++ks) {
        BF16Frag bf;
        bf.u[0] = *(const uint4*)(kp + ks * 32 + half * 8);
        bf.u[1] = *(const uint4*)(kp + ks * 32 + 16 + half * 8);
        sc[ni] = wmma_bf16(qf[ks].v, bf.v, sc[ni]);
      }
    }

    // ---- online softmax; row = r + 8*half lives in one 16-lane half ----
    float alpha[8];
    #pragma unroll
    for (int r = 0; r < 8; ++r) {
      float v = fmaxf(fmaxf(sc[0][r], sc[1][r]), fmaxf(sc[2][r], sc[3][r]));
      #pragma unroll
      for (int off = 8; off >= 1; off >>= 1) v = fmaxf(v, __shfl_xor(v, off, 32));
      float mn = fmaxf(mrow[r], v);
      alpha[r] = __expf(mrow[r] - mn);
      float s = 0.f;
      #pragma unroll
      for (int ni = 0; ni < 4; ++ni) {
        float e = __expf(sc[ni][r] - mn);
        sc[ni][r] = e;
        s += e;
      }
      #pragma unroll
      for (int off = 8; off >= 1; off >>= 1) s += __shfl_xor(s, off, 32);
      lrow[r] = lrow[r] * alpha[r] + s;
      mrow[r] = mn;
    }

    // ---- rescale O, spill P (bf16) to per-wave LDS in A layout ----
    #pragma unroll
    for (int ni = 0; ni < 4; ++ni)
      #pragma unroll
      for (int r = 0; r < 8; ++r) {
        acc_o[ni][r] *= alpha[r];
        P[wid][r + half * 8][ni * 16 + l15] = f2bf(sc[ni][r]);
      }

    // ---- O += P * V ----
    #pragma unroll
    for (int ks = 0; ks < 2; ++ks) {
      BF16Frag af;
      af.u[0] = *(const uint4*)&P[wid][l15][ks * 32 + half * 8];
      af.u[1] = *(const uint4*)&P[wid][l15][ks * 32 + 16 + half * 8];
      #pragma unroll
      for (int nv = 0; nv < 4; ++nv) {
        const unsigned short* vp =
            vbase + (size_t)(nv * 16 + l15) * S_ + kt * 64 + ks * 32 + half * 8;
        BF16Frag bf;
        bf.u[0] = *(const uint4*)vp;
        bf.u[1] = *(const uint4*)(vp + 16);
        acc_o[nv] = wmma_bf16(af.v, bf.v, acc_o[nv]);
      }
    }
  }

  #pragma unroll
  for (int nv = 0; nv < 4; ++nv)
    #pragma unroll
    for (int r = 0; r < 8; ++r) {
      int row = qt * 64 + wid * 16 + r + half * 8;
      int col = h * HD_ + nv * 16 + l15;
      AOb[((size_t)b * S_ + row) * E_ + col] = f2bf(acc_o[nv][r] / lrow[r]);
    }
}

// ---------------- sparse band path (writes d_out) ----------------
__global__ __launch_bounds__(256)
void sparse_band(const float* __restrict__ Q, const float* __restrict__ K,
                 const float* __restrict__ V, const float* __restrict__ Vsum,
                 float* __restrict__ out) {
  const int q = blockIdx.x, b = blockIdx.y;
  const int tid = threadIdx.x, lane = tid & 31, wid = tid >> 5;

  __shared__ float qrow[E_];
  __shared__ float sc[32];
  __shared__ float prob[32];
  __shared__ float zsh[2];

  const float* qp = Q + ((size_t)b * S_ + q) * E_;
  for (int e = tid; e < E_; e += 256) qrow[e] = qp[e];
  __syncthreads();

  int j0 = q - WIN_; if (j0 < 0) j0 = 0;
  int j1 = q + WIN_; if (j1 > S_ - 1) j1 = S_ - 1;
  int nb = j1 - j0 + 1;

  for (int slot = wid; slot < nb; slot += 8) {
    const float* kp = K + ((size_t)b * S_ + j0 + slot) * E_;
    float d = 0.f;
    for (int e = lane; e < E_; e += 32) d += qrow[e] * kp[e];
    #pragma unroll
    for (int off = 16; off >= 1; off >>= 1) d += __shfl_xor(d, off, 32);
    if (lane == 0) sc[slot] = d * 0.03125f;   // E^-0.5
  }
  __syncthreads();
  if (tid == 0) {
    float m = 0.f;                            // out-of-band score is exactly 0
    for (int j = 0; j < nb; ++j) m = fmaxf(m, sc[j]);
    float p0 = __expf(-m);
    float Z = (float)(S_ - nb) * p0;
    for (int j = 0; j < nb; ++j) {
      float p = __expf(sc[j] - m);
      prob[j] = p - p0;
      Z += p;
    }
    zsh[0] = 1.f / Z; zsh[1] = p0;
  }
  __syncthreads();
  float invZ = zsh[0], p0 = zsh[1];

  int e = tid * 4;
  float4 a = *(const float4*)(Vsum + b * E_ + e);
  a.x *= p0; a.y *= p0; a.z *= p0; a.w *= p0;
  for (int j = 0; j < nb; ++j) {
    float w = prob[j];
    float4 vv = *(const float4*)(V + ((size_t)b * S_ + j0 + j) * E_ + e);
    a.x += w * vv.x; a.y += w * vv.y; a.z += w * vv.z; a.w += w * vv.w;
  }
  a.x *= invZ; a.y *= invZ; a.z *= invZ; a.w *= invZ;
  *(float4*)(out + ((size_t)b * S_ + q) * E_ + e) = a;
}

// ---------------- out-projection GEMM, epilogue: out += 0.1*(C + b) ----------------
__global__ __launch_bounds__(256)
void outproj_gemm(const unsigned short* __restrict__ AOb,  // [B*S,E] bf16
                  const unsigned short* __restrict__ Wob,  // [E,E] bf16
                  const float* __restrict__ bias,          // [E]
                  float* __restrict__ out) {
  const int m0 = blockIdx.x * 128;
  const int n0 = blockIdx.y * 128;

  __shared__ unsigned short At[128][LDST];
  __shared__ unsigned short Bt[128][LDST];

  const int tid = threadIdx.x;
  const int lane = tid & 31, wid = tid >> 5;
  const int wm = (wid >> 1) * 32, wn = (wid & 1) * 64;
  const int half = lane >> 4, l15 = lane & 15;
  const int lrow = tid >> 1, lcol = (tid & 1) * 16;

  v8f acc[2][4];
  #pragma unroll
  for (int mi = 0; mi < 2; ++mi)
    #pragma unroll
    for (int ni = 0; ni < 4; ++ni) acc[mi][ni] = vzero8();

  for (int k0 = 0; k0 < E_; k0 += 32) {
    const uint4* ga = (const uint4*)(AOb + (size_t)(m0 + lrow) * E_ + k0 + lcol);
    uint4 a0 = ga[0], a1 = ga[1];
    const uint4* gb = (const uint4*)(Wob + (size_t)(n0 + lrow) * E_ + k0 + lcol);
    uint4 b0 = gb[0], b1 = gb[1];
    if (k0 + 32 < E_) {
      __builtin_prefetch(AOb + (size_t)(m0 + lrow) * E_ + k0 + 32 + lcol, 0, 3);
      __builtin_prefetch(Wob + (size_t)(n0 + lrow) * E_ + k0 + 32 + lcol, 0, 3);
    }
    __syncthreads();
    *(uint4*)&At[lrow][lcol]     = a0;
    *(uint4*)&At[lrow][lcol + 8] = a1;
    *(uint4*)&Bt[lrow][lcol]     = b0;
    *(uint4*)&Bt[lrow][lcol + 8] = b1;
    __syncthreads();

    BF16Frag afr[2], bfr[4];
    #pragma unroll
    for (int mi = 0; mi < 2; ++mi) {
      int r = wm + mi * 16 + l15;
      afr[mi].u[0] = *(const uint4*)&At[r][half * 8];
      afr[mi].u[1] = *(const uint4*)&At[r][16 + half * 8];
    }
    #pragma unroll
    for (int ni = 0; ni < 4; ++ni) {
      int c = wn + ni * 16 + l15;
      bfr[ni].u[0] = *(const uint4*)&Bt[c][half * 8];
      bfr[ni].u[1] = *(const uint4*)&Bt[c][16 + half * 8];
    }
    #pragma unroll
    for (int mi = 0; mi < 2; ++mi)
      #pragma unroll
      for (int ni = 0; ni < 4; ++ni)
        acc[mi][ni] = wmma_bf16(afr[mi].v, bfr[ni].v, acc[mi][ni]);
  }

  #pragma unroll
  for (int mi = 0; mi < 2; ++mi)
    #pragma unroll
    for (int ni = 0; ni < 4; ++ni) {
      int col = n0 + wn + ni * 16 + l15;
      float bc = bias[col];
      #pragma unroll
      for (int r = 0; r < 8; ++r) {
        int row = m0 + wm + mi * 16 + r + half * 8;
        float* p = out + (size_t)row * E_ + col;
        *p = *p + 0.1f * (acc[mi][ni][r] + bc);
      }
    }
}

// ---------------- launcher ----------------
extern "C" void kernel_launch(void* const* d_in, const int* in_sizes, int n_in,
                              void* d_out, int out_size, void* d_ws, size_t ws_size,
                              hipStream_t stream) {
  const float* Q  = (const float*)d_in[0];
  const float* K  = (const float*)d_in[1];
  const float* V  = (const float*)d_in[2];
  const float* Wi = (const float*)d_in[3];
  const float* bi = (const float*)d_in[4];
  const float* Wo = (const float*)d_in[5];
  const float* bo = (const float*)d_in[6];
  float* out = (float*)d_out;

  char* ws = (char*)d_ws;
  size_t off = 0;
  auto alloc = [&](size_t bytes) -> void* {
    void* p = ws + off;
    off += (bytes + 255) & ~(size_t)255;
    return p;
  };
  const size_t NTOK = (size_t)B_ * S_;                 // 8192 tokens
  unsigned short* Qb  = (unsigned short*)alloc(NTOK * E_ * 2);
  unsigned short* Kb  = (unsigned short*)alloc(NTOK * E_ * 2);
  unsigned short* Vb  = (unsigned short*)alloc(NTOK * E_ * 2);
  unsigned short* Wb  = (unsigned short*)alloc((size_t)3 * E_ * E_ * 2);
  unsigned short* Wob = (unsigned short*)alloc((size_t)E_ * E_ * 2);
  unsigned short* QPb = (unsigned short*)alloc(NTOK * E_ * 2);
  unsigned short* KPb = (unsigned short*)alloc(NTOK * E_ * 2);
  unsigned short* VTb = (unsigned short*)alloc(NTOK * E_ * 2);
  unsigned short* AOb = (unsigned short*)alloc(NTOK * E_ * 2);
  float* Vsum = (float*)alloc((size_t)B_ * E_ * 4);

  int nTok = (int)(NTOK * E_);
  to_bf16_kernel<<<(nTok + 255) / 256, 256, 0, stream>>>(Q, Qb, nTok);
  to_bf16_kernel<<<(nTok + 255) / 256, 256, 0, stream>>>(K, Kb, nTok);
  to_bf16_kernel<<<(nTok + 255) / 256, 256, 0, stream>>>(V, Vb, nTok);
  int nWi = 3 * E_ * E_;
  to_bf16_kernel<<<(nWi + 255) / 256, 256, 0, stream>>>(Wi, Wb, nWi);
  int nWo = E_ * E_;
  to_bf16_kernel<<<(nWo + 255) / 256, 256, 0, stream>>>(Wo, Wob, nWo);

  vsum_kernel<<<dim3(E_ / 256, B_), 256, 0, stream>>>(V, Vsum);

  inproj_gemm<<<dim3(NTOK / 128, E_ / 128, 3), 256, 0, stream>>>(
      Qb, Kb, Vb, Wb, bi, QPb, KPb, VTb);

  flash_attn<<<dim3(S_ / 64, H_, B_), 128, 0, stream>>>(QPb, KPb, VTb, AOb);

  sparse_band<<<dim3(S_, B_), 256, 0, stream>>>(Q, K, V, Vsum, out);

  outproj_gemm<<<dim3(NTOK / 128, E_ / 128), 256, 0, stream>>>(AOb, Wob, bo, out);
}
